// LinearSSM_12893491823419
// MI455X (gfx1250) — matrix operand order repaired and checked by reading, not compiled, not run on hardware
//
#include <hip/hip_runtime.h>
#include <hip/hip_bf16.h>

typedef __attribute__((ext_vector_type(16))) __bf16 v16bf;
typedef __attribute__((ext_vector_type(8)))  float  v8f;

#define BSZ    32
#define SSZ    4096
#define NSZ    512
#define HSZ    256
#define TCHUNK 256                  // timesteps per grid-z chunk (even!)
#define NCHUNK (SSZ / TCHUNK)       // 16
#define SUBT   (TCHUNK / 16)        // 16 WMMA t-subtiles per chunk

// ---------------------------------------------------------------------------
// Kernel 1: convert x (f32, row-major [B][S][N]) into bf16, pre-swizzled into
// the CDNA5 16-bit A-matrix fragment layout:
//   dword index d = ((((b*256 + tt)*16 + kb)*32 + lane)*8 + v)
//   holds bf16 pair (n, n+1) for row t = tt*16 + lane%16,
//   n = kb*32 + (v<4 ? 2v : 16+2(v-4)) + (lane>=16 ? 8 : 0)
// Fragments are contiguous 1KB blocks in flat (tt,kb) order -> the GEMM can
// stream them with a flat software pipeline.
// ---------------------------------------------------------------------------
__global__ __launch_bounds__(256) void cvt_x_kernel(const float* __restrict__ x,
                                                    unsigned* __restrict__ xb) {
  long long d = (long long)blockIdx.x * 256 + threadIdx.x;
  int v  = (int)(d & 7);
  int L  = (int)((d >> 3) & 31);
  int kb = (int)((d >> 8) & 15);
  long long rest = d >> 12;              // b*(S/16) + tt
  int tt = (int)(rest & 255);
  int b  = (int)(rest >> 8);
  int t  = tt * 16 + (L & 15);
  int noff = ((v < 4) ? (2 * v) : (16 + 2 * (v - 4))) + ((L >= 16) ? 8 : 0);
  int n = kb * 32 + noff;
  const float2* rp = (const float2*)(x + ((long long)b * SSZ + t) * NSZ + n);
  float2 f = *rp;
  // round-to-nearest bf16 (half-up) and pack two into one dword via v_perm
  unsigned u0 = __float_as_uint(f.x) + 0x8000u;
  unsigned u1 = __float_as_uint(f.y) + 0x8000u;
  xb[d] = __builtin_amdgcn_perm(u1, u0, 0x07060302);
}

// ---------------------------------------------------------------------------
// Kernel 2: M = B_w @ W_in  (512x512x512), output bf16 row-major.
// Simple LDS-tiled SGEMM; off the critical path (0.27 GFLOP).
// ---------------------------------------------------------------------------
__global__ __launch_bounds__(256) void fuse_w_kernel(const float* __restrict__ Bw,
                                                     const float* __restrict__ Win,
                                                     unsigned short* __restrict__ Mb) {
  __shared__ float sB[16][16];
  __shared__ float sW[16][17];
  int tx = threadIdx.x, ty = threadIdx.y;
  int m = blockIdx.y * 16 + ty;
  int n = blockIdx.x * 16 + tx;
  float acc = 0.f;
  for (int k0 = 0; k0 < NSZ; k0 += 16) {
    sB[ty][tx] = Bw[m * NSZ + k0 + tx];
    sW[ty][tx] = Win[(k0 + ty) * NSZ + n];
    __syncthreads();
#pragma unroll
    for (int k = 0; k < 16; ++k) acc += sB[ty][k] * sW[k][tx];
    __syncthreads();
  }
  unsigned u = __float_as_uint(acc) + 0x8000u;
  Mb[m * NSZ + n] = (unsigned short)(u >> 16);
}

// ---------------------------------------------------------------------------
// Kernel 3: fused GEMM + weighted time-scan, software-pipelined.
// One wave per (batch, 16-wide m-tile, time-chunk).
// B fragments (16 x 8 VGPRs) resident for the whole chunk; A fragments stream
// through a 4-slot modular buffer with prefetch distance 3 (pipeline crosses
// subtile boundaries since fragments are flat-contiguous). Two independent
// WMMA accumulation chains avoid back-to-back C-dependency stalls.
// ---------------------------------------------------------------------------
__global__ __launch_bounds__(32) void ssm_gemm_kernel(const unsigned* __restrict__ xb,
                                                      const unsigned short* __restrict__ Mb,
                                                      const float* __restrict__ Avec,
                                                      float* __restrict__ partial) {
  const int lane = threadIdx.x;
  const int b  = blockIdx.x;
  const int mt = blockIdx.y;
  const int z  = blockIdx.z;
  const int m  = mt * 16 + (lane & 15);

  // --- load 16 B fragments (K-blocks of 32) for this m-tile, keep resident ---
  v16bf bfrag[16];
  {
    const unsigned short* row = Mb + (long long)m * NSZ + ((lane >= 16) ? 8 : 0);
#pragma unroll
    for (int kb = 0; kb < 16; ++kb) {
      union { uint4 q[2]; v16bf v; } u;
      const uint4* p = (const uint4*)(row + kb * 32);
      u.q[0] = p[0];
      u.q[1] = p[2];
      bfrag[kb] = u.v;
    }
  }

  // --- per-channel decay powers ---
  float a = Avec[m];
  float w[8];                 // w[v] = a^(7-v)
  w[7] = 1.f;
#pragma unroll
  for (int i = 6; i >= 0; --i) w[i] = w[i + 1] * a;
  float a8  = w[0] * a;
  float a16 = a8 * a8;

  // --- A-fragment stream: chunk base; fragment f at xt[f*64], xt[f*64+1] ---
  union AB { uint4 q[2]; v16bf v; };
  AB buf[4];
  const uint4* xt =
      (const uint4*)(xb + ((long long)(b * 256 + z * SUBT) * 16) * 256) + lane * 2;
#pragma unroll
  for (int i = 0; i < 3; ++i) {        // prime the pipeline (fragments 0..2)
    buf[i].q[0] = xt[i * 64];
    buf[i].q[1] = xt[i * 64 + 1];
  }

  float state = 0.f;

  for (int s = 0; s < SUBT; ++s) {
    const uint4* xs = xt + s * 16 * 64;
    v8f acc0 = {};
    v8f acc1 = {};
#pragma unroll
    for (int kb = 0; kb < 16; ++kb) {
      // prefetch fragment f+3 (over-reads <=3KB past final chunk; lands in
      // the Mb/partial workspace region, still in-bounds)
      buf[(kb + 3) & 3].q[0] = xs[(kb + 3) * 64];
      buf[(kb + 3) & 3].q[1] = xs[(kb + 3) * 64 + 1];
      if ((kb & 1) == 0)
        acc0 = __builtin_amdgcn_wmma_f32_16x16x32_bf16(
            false, buf[kb & 3].v, false, bfrag[kb], (short)0, acc0, false, false);
      else
        acc1 = __builtin_amdgcn_wmma_f32_16x16x32_bf16(
            false, buf[kb & 3].v, false, bfrag[kb], (short)0, acc1, false, false);
    }
    v8f acc = acc0 + acc1;
    // fold 16x16 tile (rows = local time i) into running state:
    // lanes 0-15 hold i = v (v=0..7), lanes 16-31 hold i = v+8, column m = lane%16
    float q = 0.f;
#pragma unroll
    for (int v = 0; v < 8; ++v) q += w[v] * acc[v];
    float other = __shfl_xor(q, 16, 32);
    float c = (lane < 16) ? (a8 * q + other) : (a8 * other + q);
    state = state * a16 + c;    // chunk recurrence at 16-step granularity
  }

  // global chunk weight A^((NCHUNK-1-z)*TCHUNK); exponent even -> sign-safe
  int e = (NCHUNK - 1 - z) * TCHUNK;
  state *= __powf(fabsf(a), (float)e);

  if (lane < 16)
    partial[((long long)z * BSZ + b) * NSZ + m] = state;
}

// ---------------------------------------------------------------------------
// Kernel 4: reduce chunk partials (+ bias geometric-sum term), C-projection,
// LayerNorm, Linear+ReLU, Linear. One 512-thread workgroup per batch row.
// ---------------------------------------------------------------------------
__global__ __launch_bounds__(512) void head_kernel(const float* __restrict__ partial,
                                                   const float* __restrict__ Bw,
                                                   const float* __restrict__ b_in,
                                                   const float* __restrict__ Avec,
                                                   const float* __restrict__ Cw,
                                                   const float* __restrict__ ln_g,
                                                   const float* __restrict__ ln_b,
                                                   const float* __restrict__ W1,
                                                   const float* __restrict__ b1,
                                                   const float* __restrict__ W2,
                                                   const float* __restrict__ b2,
                                                   float* __restrict__ out) {
  __shared__ float s_state[NSZ];
  __shared__ float s_o[NSZ];
  __shared__ float red[512];
  const int b = blockIdx.x;
  const int tid = threadIdx.x;

  { // state[m] = sum_z partial[z][b][m] + (B_w b_in)[m] * geom(A[m])
    float s = 0.f;
    for (int zz = 0; zz < NCHUNK; ++zz)
      s += partial[((long long)zz * BSZ + b) * NSZ + tid];
    float c = 0.f;
    for (int k = 0; k < NSZ; ++k) c += Bw[(long long)tid * NSZ + k] * b_in[k];
    float a = Avec[tid];
    float aS = __powf(fabsf(a), (float)SSZ);        // S even
    float geom = (fabsf(1.f - a) > 1e-12f) ? (1.f - aS) / (1.f - a) : (float)SSZ;
    s_state[tid] = s + c * geom;
  }
  __syncthreads();

  { // o = state @ C_w^T
    float acc = 0.f;
    for (int mm = 0; mm < NSZ; ++mm) acc += Cw[(long long)tid * NSZ + mm] * s_state[mm];
    s_o[tid] = acc;
  }
  __syncthreads();

  // LayerNorm over 512
  float v = s_o[tid];
  red[tid] = v;
  __syncthreads();
  for (int off = 256; off > 0; off >>= 1) {
    if (tid < off) red[tid] += red[tid + off];
    __syncthreads();
  }
  float mu = red[0] / (float)NSZ;
  __syncthreads();
  float dv = v - mu;
  red[tid] = dv * dv;
  __syncthreads();
  for (int off = 256; off > 0; off >>= 1) {
    if (tid < off) red[tid] += red[tid + off];
    __syncthreads();
  }
  float inv = rsqrtf(red[0] / (float)NSZ + 1e-5f);
  __syncthreads();
  s_o[tid] = (v - mu) * inv * ln_g[tid] + ln_b[tid];
  __syncthreads();

  // relu(o @ W1^T + b1) .* W2 row
  if (tid < HSZ) {
    float acc = b1[tid];
    for (int n = 0; n < NSZ; ++n) acc += W1[(long long)tid * NSZ + n] * s_o[n];
    red[tid] = fmaxf(acc, 0.f) * W2[tid];
  }
  __syncthreads();
  for (int off = 128; off > 0; off >>= 1) {
    if (tid < off) red[tid] += red[tid + off];
    __syncthreads();
  }
  if (tid == 0) out[b] = red[0] + b2[0];
}

// ---------------------------------------------------------------------------
extern "C" void kernel_launch(void* const* d_in, const int* in_sizes, int n_in,
                              void* d_out, int out_size, void* d_ws, size_t ws_size,
                              hipStream_t stream) {
  const float* x    = (const float*)d_in[0];
  const float* W_in = (const float*)d_in[1];
  const float* b_in = (const float*)d_in[2];
  const float* A    = (const float*)d_in[3];
  const float* B_w  = (const float*)d_in[4];
  const float* C_w  = (const float*)d_in[5];
  const float* ln_g = (const float*)d_in[6];
  const float* ln_b = (const float*)d_in[7];
  const float* W1   = (const float*)d_in[8];
  const float* b1   = (const float*)d_in[9];
  const float* W2   = (const float*)d_in[10];
  const float* b2   = (const float*)d_in[11];
  float* out = (float*)d_out;

  char* ws = (char*)d_ws;
  unsigned* xb = (unsigned*)ws;                              // 134,217,728 B
  size_t xb_bytes = (size_t)BSZ * SSZ * NSZ * 2;
  unsigned short* Mb = (unsigned short*)(ws + xb_bytes);     // 524,288 B
  size_t mb_bytes = (size_t)NSZ * NSZ * 2;
  float* partial = (float*)(ws + xb_bytes + mb_bytes);       // 1,048,576 B

  long long tot_dwords = (long long)BSZ * SSZ * NSZ / 2;     // 33,554,432
  cvt_x_kernel<<<dim3((unsigned)(tot_dwords / 256)), 256, 0, stream>>>(x, xb);
  fuse_w_kernel<<<dim3(NSZ / 16, NSZ / 16), dim3(16, 16), 0, stream>>>(B_w, W_in, Mb);
  ssm_gemm_kernel<<<dim3(BSZ, NSZ / 16, NCHUNK), 32, 0, stream>>>(xb, Mb, A, partial);
  head_kernel<<<dim3(BSZ), 512, 0, stream>>>(partial, B_w, b_in, A, C_w, ln_g, ln_b,
                                             W1, b1, W2, b2, out);
}
